// LlamaMoDDecoderLayer_17162689315243
// MI455X (gfx1250) — compile-verified, matrix operand order checked by reading, not run on hardware
//
#include <hip/hip_runtime.h>

typedef __bf16 bf16_t;
typedef bf16_t bf16x16 __attribute__((ext_vector_type(16)));
typedef float  f32x8   __attribute__((ext_vector_type(8)));

#define S_LEN 2048
#define D_DIM 2048
#define H_NUM 16
#define DH    128
#define F_DIM 8192

// ---------------- helpers ----------------

// native f32->bf16 convert (backend picks v_cvt*_bf16_f32)
__device__ __forceinline__ unsigned short f2bf(float x){
  union { bf16_t h; unsigned short u; } t;
  t.h = (bf16_t)x;
  return t.u;
}

// pack two floats into a bf16 pair (lo in [15:0], hi in [31:16])
__device__ __forceinline__ unsigned pack_bf2(float lo, float hi){
  union { bf16_t h[2]; unsigned u; } t;
  t.h[0] = (bf16_t)lo;
  t.h[1] = (bf16_t)hi;
  return t.u;
}

// fast reciprocal -> v_rcp_f32 (1 ULP, fine for activations)
__device__ __forceinline__ float fast_rcp(float x){
  return __builtin_amdgcn_rcpf(x);
}

// Build a 16xbf16 WMMA fragment from LDS. `base` points at the lane's row
// (pitch keeps it 16B aligned); koff = (lane>>4)*8 selects the half-wave's
// K offset. Two ds_load_b128: VGPRs 0..3 hold K=koff..koff+7, VGPRs 4..7 hold
// K=16+koff..16+koff+7 (per CDNA5 16-bit A/B layout).
__device__ __forceinline__ bf16x16 load_frag(const unsigned short* base, int koff){
  union { bf16x16 f; uint4 u[2]; } r;
  r.u[0] = *reinterpret_cast<const uint4*>(base + koff);
  r.u[1] = *reinterpret_cast<const uint4*>(base + koff + 16);
  return r.f;
}

__device__ __forceinline__ f32x8 wmma_bf16(bf16x16 a, bf16x16 b, f32x8 c){
  return __builtin_amdgcn_wmma_f32_16x16x32_bf16(false, a, false, b, (short)0, c,
                                                 false, false);
}

// async copy 16 bytes global->LDS (CDNA5 GV mode), tracked by ASYNCcnt
__device__ __forceinline__ void async_copy_b128(unsigned lds_off, const void* gptr){
  unsigned long long ga = (unsigned long long)gptr;
  asm volatile("global_load_async_to_lds_b128 %0, %1, off"
               :: "v"(lds_off), "v"(ga) : "memory");
}

__device__ __forceinline__ float block_sum256(float v, float* sbuf){
  #pragma unroll
  for (int off = 16; off >= 1; off >>= 1) v += __shfl_xor(v, off, 32);
  int w = threadIdx.x >> 5, ln = threadIdx.x & 31;
  __syncthreads();
  if (ln == 0) sbuf[w] = v;
  __syncthreads();
  float r = 0.f;
  #pragma unroll
  for (int i = 0; i < 8; i++) r += sbuf[i];
  return r;
}

// ---------------- kernel 1: RMSNorm(ln1) + routers ----------------

__global__ __launch_bounds__(256)
void prep_kernel(const float* __restrict__ hs, const float* __restrict__ ln1,
                 const float* __restrict__ raw, const float* __restrict__ rab,
                 const float* __restrict__ rmw, const float* __restrict__ rmb,
                 unsigned short* __restrict__ xnbf,
                 unsigned char* __restrict__ mattn, unsigned char* __restrict__ mmlp)
{
  __shared__ float sbuf[8];
  int s = blockIdx.x, tid = threadIdx.x;
  const float* row = hs + (size_t)s * D_DIM;
  float ss = 0.f, a0 = 0.f, a1 = 0.f, m0 = 0.f, m1 = 0.f;
  for (int i = tid; i < D_DIM; i += 256){
    float x = row[i];
    ss += x * x;
    a0 += x * raw[i*2+0]; a1 += x * raw[i*2+1];
    m0 += x * rmw[i*2+0]; m1 += x * rmw[i*2+1];
  }
  ss = block_sum256(ss, sbuf);
  a0 = block_sum256(a0, sbuf);
  a1 = block_sum256(a1, sbuf);
  m0 = block_sum256(m0, sbuf);
  m1 = block_sum256(m1, sbuf);
  float r = rsqrtf(ss * (1.0f / D_DIM) + 1e-5f);
  for (int i = tid * 2; i < D_DIM; i += 512){
    float y0 = row[i]   * r * ln1[i];
    float y1 = row[i+1] * r * ln1[i+1];
    *reinterpret_cast<unsigned*>(&xnbf[(size_t)s * D_DIM + i]) = pack_bf2(y0, y1);
  }
  if (tid == 0){
    mattn[s] = ((a1 + rab[1]) > (a0 + rab[0])) ? 1 : 0;   // argmax==1
    mmlp[s]  = ((m1 + rmb[1]) > (m0 + rmb[0])) ? 1 : 0;
  }
}

// ---------------- kernel 2: double-buffered WMMA GEMM ----------------
// C[M,N] = A(bf16,[M,K]) @ B(f32,[K,N]); BM=BN=128, BK=32; 8 waves, each 32x64.
// Pipeline: while tile i is consumed by 8 WMMAs, tile i+1's A is DMA'd via
// global_load_async_to_lds_b128 and B is converted/transposed into LDS; the
// wave drains ASYNCcnt only after the math, then barriers.
// mode 0: store f32 C. mode 1: outbf = bf16( silu(gateb) * acc ).

__global__ __launch_bounds__(256)
void gemm_bf16(const unsigned short* __restrict__ A, const float* __restrict__ B,
               float* __restrict__ C, const float* __restrict__ gateb,
               unsigned short* __restrict__ outbf,
               int M, int N, int K, int mode)
{
  __shared__ unsigned short As[2][128 * 32];   // [m][k]
  __shared__ unsigned short Bs[2][128 * 32];   // transposed: [n][k]
  const int tid = threadIdx.x;
  const int m0 = blockIdx.y * 128, n0 = blockIdx.x * 128;
  const int wid = tid >> 5, lane = tid & 31;
  const int wm = wid >> 1, wn = wid & 1;
  const int half = lane >> 4, ln16 = lane & 15;
  const int koff = half * 8;

  f32x8 zero = {0.f,0.f,0.f,0.f,0.f,0.f,0.f,0.f};
  f32x8 acc[2][4];
  #pragma unroll
  for (int a = 0; a < 2; a++)
    #pragma unroll
    for (int b = 0; b < 4; b++) acc[a][b] = zero;

  // per-thread staging coordinates
  const int arow = tid >> 2, ac4 = tid & 3;            // A: 2 x (row, 16B chunk)
  const int bkp  = tid >> 5, bc4 = tid & 31;           // B: 2 x (k-pair, n-chunk)

  auto stage = [&](int buf, int k0){
    // A: 512 async b128 copies (bf16, no conversion, no VGPR round-trip)
    #pragma unroll
    for (int i = 0; i < 2; i++){
      int row = arow + i * 64;
      unsigned lds = (unsigned)(size_t)&As[buf][row * 32 + ac4 * 8];
      async_copy_b128(lds, A + (size_t)(m0 + row) * K + k0 + ac4 * 8);
    }
    // B: two adjacent K-rows as float4 -> packed bf16 pairs, one b32 store each
    #pragma unroll
    for (int i = 0; i < 2; i++){
      int kk = (bkp + i * 8) * 2;
      const float4 fa = *reinterpret_cast<const float4*>(B + (size_t)(k0 + kk    ) * N + n0 + bc4 * 4);
      const float4 fb = *reinterpret_cast<const float4*>(B + (size_t)(k0 + kk + 1) * N + n0 + bc4 * 4);
      int nb = bc4 * 4;
      *reinterpret_cast<unsigned*>(&Bs[buf][(nb + 0) * 32 + kk]) = pack_bf2(fa.x, fb.x);
      *reinterpret_cast<unsigned*>(&Bs[buf][(nb + 1) * 32 + kk]) = pack_bf2(fa.y, fb.y);
      *reinterpret_cast<unsigned*>(&Bs[buf][(nb + 2) * 32 + kk]) = pack_bf2(fa.z, fb.z);
      *reinterpret_cast<unsigned*>(&Bs[buf][(nb + 3) * 32 + kk]) = pack_bf2(fa.w, fb.w);
    }
    // speculative prefetch of the tile after this one -> global_prefetch_b8
    if (k0 + 32 < K)
      __builtin_prefetch(B + (size_t)(k0 + 32 + (tid & 31)) * N + n0 + (tid >> 5) * 16, 0, 0);
  };

  // prologue: stage tile 0, publish
  stage(0, 0);
  asm volatile("s_wait_asynccnt 0" ::: "memory");
  __syncthreads();

  const int nk = K >> 5;
  for (int it = 0; it < nk; ++it){
    const int cur = it & 1;
    if (it + 1 < nk) stage(cur ^ 1, (it + 1) << 5);   // overlap with WMMAs below

    bf16x16 af[2], bfm[4];
    #pragma unroll
    for (int am = 0; am < 2; am++)
      af[am] = load_frag(&As[cur][(wm * 32 + am * 16 + ln16) * 32], koff);
    #pragma unroll
    for (int bn = 0; bn < 4; bn++)
      bfm[bn] = load_frag(&Bs[cur][(wn * 64 + bn * 16 + ln16) * 32], koff);
    #pragma unroll
    for (int am = 0; am < 2; am++)
      #pragma unroll
      for (int bn = 0; bn < 4; bn++)
        acc[am][bn] = wmma_bf16(af[am], bfm[bn], acc[am][bn]);

    if (it + 1 < nk){
      asm volatile("s_wait_asynccnt 0" ::: "memory");  // next A tile landed
      __syncthreads();                                 // publish next tile
    }
  }

  #pragma unroll
  for (int am = 0; am < 2; am++){
    #pragma unroll
    for (int bn = 0; bn < 4; bn++){
      int mg = m0 + wm * 32 + am * 16 + 8 * half;
      int ng = n0 + wn * 64 + bn * 16 + ln16;
      #pragma unroll
      for (int r = 0; r < 8; r++){
        size_t idx = (size_t)(mg + r) * N + ng;
        float v = acc[am][bn][r];
        if (mode == 0){
          C[idx] = v;
        } else {
          float g = gateb[idx];
          float h = g * fast_rcp(1.f + __expf(-g)) * v;   // silu(g)*up, v_rcp_f32
          outbf[idx] = f2bf(h);
        }
      }
    }
  }
}

// ---------------- kernel 3: RoPE (in-place on q,k) ----------------

__global__ __launch_bounds__(256)
void rope_kernel(float* __restrict__ q, float* __restrict__ k)
{
  int idx = blockIdx.x * 256 + threadIdx.x;    // S*H*64 threads
  int j = idx & 63;
  int h = (idx >> 6) & (H_NUM - 1);
  int s = idx >> 10;
  if (s >= S_LEN) return;
  float inv = __powf(10000.f, -(float)j * (1.f / 64.f));
  float ang = (float)s * inv;
  float sn, cs; __sincosf(ang, &sn, &cs);
  size_t base = (size_t)s * D_DIM + h * DH;
  float q0 = q[base + j], q1 = q[base + j + 64];
  q[base + j]      = q0 * cs - q1 * sn;
  q[base + j + 64] = q1 * cs + q0 * sn;
  float k0 = k[base + j], k1 = k[base + j + 64];
  k[base + j]      = k0 * cs - k1 * sn;
  k[base + j + 64] = k1 * cs + k0 * sn;
}

// ---------------- kernel 4: flash attention (WMMA, online softmax) ----------------
// grid (S/64, H), 128 threads = 4 waves, wave w owns query rows q0+16w..+15.

__global__ __launch_bounds__(128)
void flash_attn(const float* __restrict__ q, const float* __restrict__ k,
                const float* __restrict__ v, const unsigned char* __restrict__ amask,
                unsigned short* __restrict__ obf)
{
  __shared__ unsigned short Qs[64 * 128];    // [q][dh]
  __shared__ unsigned short Ks[32 * 128];    // [key][dh]
  __shared__ unsigned short Vst[128 * 32];   // transposed: [dh][key]
  __shared__ unsigned short Ps[4 * 16 * 32]; // per-wave P tile [q][key]
  const int tid = threadIdx.x, lane = tid & 31, w = tid >> 5;
  const int half = lane >> 4, ln16 = lane & 15;
  const int koff = half * 8;
  const int h = blockIdx.y, q0 = blockIdx.x * 64;

  // stage Q (64x128 f32 -> packed bf16, ds_store_b64)
  #pragma unroll
  for (int i = 0; i < 16; i++){
    int id = tid + i * 128;
    int row = id >> 5, c4 = id & 31;
    float4 f = *reinterpret_cast<const float4*>(q + (size_t)(q0 + row) * D_DIM + h * DH + c4 * 4);
    uint2 t; t.x = pack_bf2(f.x, f.y); t.y = pack_bf2(f.z, f.w);
    *reinterpret_cast<uint2*>(&Qs[row * 128 + c4 * 4]) = t;
  }

  f32x8 zero = {0.f,0.f,0.f,0.f,0.f,0.f,0.f,0.f};
  f32x8 o[8];
  #pragma unroll
  for (int d8 = 0; d8 < 8; d8++) o[d8] = zero;
  float m_r[8], l_r[8];
  #pragma unroll
  for (int r = 0; r < 8; r++){ m_r[r] = -3.0e38f; l_r[r] = 0.f; }
  const float scale = 0.08838834764831845f;   // 1/sqrt(128)

  const int kend = q0 + 64;                   // causal upper bound for this block
  for (int kb = 0; kb < kend; kb += 32){
    __syncthreads();
    // stage K natural (packed pairs along dh)
    #pragma unroll
    for (int i = 0; i < 8; i++){
      int id = tid + i * 128;
      int row = id >> 5, c4 = id & 31;
      float4 fk = *reinterpret_cast<const float4*>(k + (size_t)(kb + row) * D_DIM + h * DH + c4 * 4);
      uint2 t; t.x = pack_bf2(fk.x, fk.y); t.y = pack_bf2(fk.z, fk.w);
      *reinterpret_cast<uint2*>(&Ks[row * 128 + c4 * 4]) = t;
    }
    // stage V transposed (pairs over adjacent keys -> one b32 per (dh, key-pair))
    #pragma unroll
    for (int i = 0; i < 4; i++){
      int id = tid + i * 128;                // 0..511
      int kp = id >> 5, c4 = id & 31;        // kp 0..15
      int kk = kp * 2;
      size_t g0 = (size_t)(kb + kk) * D_DIM + h * DH + c4 * 4;
      float4 a = *reinterpret_cast<const float4*>(v + g0);
      float4 b = *reinterpret_cast<const float4*>(v + g0 + D_DIM);
      int c = c4 * 4;
      *reinterpret_cast<unsigned*>(&Vst[(c + 0) * 32 + kk]) = pack_bf2(a.x, b.x);
      *reinterpret_cast<unsigned*>(&Vst[(c + 1) * 32 + kk]) = pack_bf2(a.y, b.y);
      *reinterpret_cast<unsigned*>(&Vst[(c + 2) * 32 + kk]) = pack_bf2(a.z, b.z);
      *reinterpret_cast<unsigned*>(&Vst[(c + 3) * 32 + kk]) = pack_bf2(a.w, b.w);
    }
    __syncthreads();

    // scores: Q(16x128) @ K^T -> two 16x16 f32 accumulators (keys kb.. and kb+16..)
    f32x8 s0 = zero, s1 = zero;
    #pragma unroll
    for (int ks = 0; ks < 4; ks++){
      bf16x16 aq = load_frag(&Qs[(w * 16 + ln16) * 128 + ks * 32], koff);
      bf16x16 b0 = load_frag(&Ks[(ln16) * 128 + ks * 32], koff);
      bf16x16 b1 = load_frag(&Ks[(16 + ln16) * 128 + ks * 32], koff);
      s0 = wmma_bf16(aq, b0, s0);
      s1 = wmma_bf16(aq, b1, s1);
    }

    const int qbase = q0 + w * 16 + 8 * half;
    const int k0i = kb + ln16, k1i = kb + 16 + ln16;
    const bool ok0 = amask[k0i] != 0, ok1 = amask[k1i] != 0;

    #pragma unroll
    for (int r = 0; r < 8; r++){
      int qi = qbase + r;
      float v0 = s0[r] * scale, v1 = s1[r] * scale;
      s0[r] = (ok0 && k0i <= qi) ? v0 : -3.0e38f;
      s1[r] = (ok1 && k1i <= qi) ? v1 : -3.0e38f;
    }

    // online softmax per row (row lives in VGPR r + half; cols across 16 lanes)
    #pragma unroll
    for (int r = 0; r < 8; r++){
      float mx = fmaxf(s0[r], s1[r]);
      #pragma unroll
      for (int off = 1; off < 16; off <<= 1) mx = fmaxf(mx, __shfl_xor(mx, off, 32));
      float mnew = fmaxf(m_r[r], mx);
      float alpha = __expf(m_r[r] - mnew);
      float p0 = __expf(s0[r] - mnew), p1 = __expf(s1[r] - mnew);
      float rs = p0 + p1;
      #pragma unroll
      for (int off = 1; off < 16; off <<= 1) rs += __shfl_xor(rs, off, 32);
      l_r[r] = l_r[r] * alpha + rs;
      m_r[r] = mnew;
      #pragma unroll
      for (int d8 = 0; d8 < 8; d8++) o[d8][r] *= alpha;
      Ps[(w * 16 + r + 8 * half) * 32 + ln16]      = f2bf(p0);
      Ps[(w * 16 + r + 8 * half) * 32 + 16 + ln16] = f2bf(p1);
    }
    asm volatile("s_wait_dscnt 0" ::: "memory");   // wave-private LDS RAW

    // O += P(16x32) @ V(32x128): 8 WMMAs over Dh sub-tiles
    bf16x16 pf = load_frag(&Ps[(w * 16 + ln16) * 32], koff);
    #pragma unroll
    for (int d8 = 0; d8 < 8; d8++){
      bf16x16 bv = load_frag(&Vst[(d8 * 16 + ln16) * 32], koff);
      o[d8] = wmma_bf16(pf, bv, o[d8]);
    }
  }

  // epilogue: normalize (one v_rcp per row) and store bf16 attn output
  float rl[8];
  #pragma unroll
  for (int r = 0; r < 8; r++) rl[r] = fast_rcp(l_r[r]);
  #pragma unroll
  for (int d8 = 0; d8 < 8; d8++){
    #pragma unroll
    for (int r = 0; r < 8; r++){
      int row = q0 + w * 16 + r + 8 * half;
      float val = o[d8][r] * rl[r];
      obf[(size_t)row * D_DIM + h * DH + d8 * 16 + ln16] = f2bf(val);
    }
  }
}

// ---------------- kernel 5: attn residual + route mask + RMSNorm(ln2) ----------------

__global__ __launch_bounds__(256)
void attn_res_norm(const float* __restrict__ hs, const float* __restrict__ aproj,
                   const float* __restrict__ ln2, const unsigned char* __restrict__ mattn,
                   float* __restrict__ hs2, unsigned short* __restrict__ xn2)
{
  __shared__ float sbuf[8];
  int s = blockIdx.x, tid = threadIdx.x;
  float keep = mattn[s] ? 0.f : 1.f;
  float ss = 0.f;
  for (int i = tid; i < D_DIM; i += 256){
    size_t idx = (size_t)s * D_DIM + i;
    float y = hs[idx] + keep * aproj[idx];
    hs2[idx] = y;
    ss += y * y;
  }
  ss = block_sum256(ss, sbuf);
  float r = rsqrtf(ss * (1.0f / D_DIM) + 1e-5f);
  for (int i = tid * 2; i < D_DIM; i += 512){
    size_t idx = (size_t)s * D_DIM + i;
    float y0 = hs2[idx]     * r * ln2[i];
    float y1 = hs2[idx + 1] * r * ln2[i + 1];
    *reinterpret_cast<unsigned*>(&xn2[idx]) = pack_bf2(y0, y1);
  }
}

// ---------------- kernel 6: final residual with MLP query mask ----------------

__global__ __launch_bounds__(256)
void final_res(const float* __restrict__ hs2, const float* __restrict__ mlp,
               const unsigned char* __restrict__ mmlp, float* __restrict__ out)
{
  size_t i = (size_t)blockIdx.x * 256 + threadIdx.x;
  int s = (int)(i / D_DIM);
  float keep = mmlp[s] ? 0.f : 1.f;           // query_mask = !route_mask
  out[i] = hs2[i] + keep * mlp[i];
}

// ---------------- host launcher ----------------

extern "C" void kernel_launch(void* const* d_in, const int* in_sizes, int n_in,
                              void* d_out, int out_size, void* d_ws, size_t ws_size,
                              hipStream_t stream)
{
  const float* hs    = (const float*)d_in[0];
  const unsigned char* amask = (const unsigned char*)d_in[1];
  const float* Wq    = (const float*)d_in[2];
  const float* Wk    = (const float*)d_in[3];
  const float* Wv    = (const float*)d_in[4];
  const float* Wo    = (const float*)d_in[5];
  const float* ln1   = (const float*)d_in[6];
  const float* ln2   = (const float*)d_in[7];
  const float* wg    = (const float*)d_in[8];
  const float* wu    = (const float*)d_in[9];
  const float* wd    = (const float*)d_in[10];
  const float* raw   = (const float*)d_in[11];
  const float* rab   = (const float*)d_in[12];
  const float* rmw   = (const float*)d_in[13];
  const float* rmb   = (const float*)d_in[14];
  (void)in_sizes; (void)n_in; (void)out_size; (void)ws_size;

  const size_t MiB = (size_t)1 << 20;
  char* ws = (char*)d_ws;
  unsigned short* xnbf   = (unsigned short*)(ws + 0);          // 8 MiB (reused as xn2)
  unsigned char*  mattn  = (unsigned char*)(ws + 8 * MiB);
  unsigned char*  mmlp   = (unsigned char*)(ws + 8 * MiB + 4096);
  float* qbuf  = (float*)(ws + 9  * MiB);                      // 16 MiB
  float* kbuf  = (float*)(ws + 25 * MiB);                      // 16 MiB
  float* vbuf  = (float*)(ws + 41 * MiB);                      // 16 MiB
  unsigned short* attnbf = (unsigned short*)(ws + 57 * MiB);   // 8 MiB
  float* hs2   = (float*)(ws + 65 * MiB);                      // 16 MiB
  float* gbuf  = (float*)(ws + 81 * MiB);                      // 64 MiB (reused: mlp_out)
  float* aproj = qbuf;                                         // reuse after flash
  unsigned short* hbf = (unsigned short*)(ws + 9 * MiB);       // 32 MiB (reuse q+k)
  float* mlpout = gbuf;                                        // reuse after silu fuse

  // 1) norm1 + routers
  prep_kernel<<<S_LEN, 256, 0, stream>>>(hs, ln1, raw, rab, rmw, rmb, xnbf, mattn, mmlp);

  // 2) Q/K/V projections (WMMA)
  dim3 gSq(D_DIM / 128, S_LEN / 128);
  gemm_bf16<<<gSq, 256, 0, stream>>>(xnbf, Wq, qbuf, nullptr, nullptr, S_LEN, D_DIM, D_DIM, 0);
  gemm_bf16<<<gSq, 256, 0, stream>>>(xnbf, Wk, kbuf, nullptr, nullptr, S_LEN, D_DIM, D_DIM, 0);
  gemm_bf16<<<gSq, 256, 0, stream>>>(xnbf, Wv, vbuf, nullptr, nullptr, S_LEN, D_DIM, D_DIM, 0);

  // 3) RoPE
  rope_kernel<<<(S_LEN * H_NUM * 64) / 256, 256, 0, stream>>>(qbuf, kbuf);

  // 4) flash attention
  flash_attn<<<dim3(S_LEN / 64, H_NUM), 128, 0, stream>>>(qbuf, kbuf, vbuf, amask, attnbf);

  // 5) O projection, residual + attn route mask + norm2
  gemm_bf16<<<gSq, 256, 0, stream>>>(attnbf, Wo, aproj, nullptr, nullptr, S_LEN, D_DIM, D_DIM, 0);
  attn_res_norm<<<S_LEN, 256, 0, stream>>>(hs, aproj, ln2, mattn, hs2, xnbf);

  // 6) MLP: gate, fused silu*up -> bf16, down
  dim3 gSf(F_DIM / 128, S_LEN / 128);
  gemm_bf16<<<gSf, 256, 0, stream>>>(xnbf, wg, gbuf, nullptr, nullptr, S_LEN, F_DIM, D_DIM, 0);
  gemm_bf16<<<gSf, 256, 0, stream>>>(xnbf, wu, nullptr, gbuf, hbf, S_LEN, F_DIM, D_DIM, 1);
  gemm_bf16<<<gSq, 256, 0, stream>>>(hbf, wd, mlpout, nullptr, nullptr, S_LEN, D_DIM, F_DIM, 0);

  // 7) final residual with MLP query mask
  final_res<<<(S_LEN * D_DIM) / 256, 256, 0, stream>>>(hs2, mlpout, mmlp, (float*)d_out);
}
